// MLAAttention_64707977282231
// MI455X (gfx1250) — matrix operand order repaired and checked by reading, not compiled, not run on hardware
//
#include <hip/hip_runtime.h>
#include <hip/hip_bf16.h>
#include <cstdint>

// ---------------- problem constants ----------------
#define B_    2
#define S_    2048
#define HID_  2048
#define H_    16
#define KV_   4
#define HD_   128
#define RD_   64
#define FULL_ 192
#define ROWS  (B_*S_)          // 4096 token rows

typedef __attribute__((ext_vector_type(16))) __bf16 v16bf;
typedef __attribute__((ext_vector_type(8)))  float  v8f;

union Frag { unsigned u[8]; v16bf v; };

// ---------------- gfx1250 async global->LDS (ASYNCcnt) ----------------
#define ASG __attribute__((address_space(1)))
#define ASL __attribute__((address_space(3)))
typedef __attribute__((__vector_size__(16))) int v4i_t;

#if defined(__has_builtin)
#if __has_builtin(__builtin_amdgcn_global_load_async_to_lds_b128) && \
    __has_builtin(__builtin_amdgcn_s_wait_asynccnt)
#define HAVE_ASYNC 1
#endif
#endif

__device__ __forceinline__ void async_b128(const void* g, void* l) {
#ifdef HAVE_ASYNC
  __builtin_amdgcn_global_load_async_to_lds_b128(
      (ASG v4i_t*)(void*)g, (ASL v4i_t*)l, 0, 0);
#else
  *(uint4*)l = *(const uint4*)g;
#endif
}
__device__ __forceinline__ void async_wait0() {
#ifdef HAVE_ASYNC
  __builtin_amdgcn_s_wait_asynccnt(0);
#endif
}

// ---------------- f32 -> bf16 conversion ----------------
__global__ void cvt_f32_bf16(const float* __restrict__ in, __bf16* __restrict__ out, int n) {
  int i = blockIdx.x * blockDim.x + threadIdx.x;
  if (i < n) out[i] = (__bf16)in[i];
}

// ---------------- generic bf16 WMMA GEMM: C[M,N] = A[M,K] @ B[K,N] ----------------
// block: 256 threads (8 waves), tile 128x128, K-step 32, double-buffered LDS.
// wave (wv>>1) -> 32-row slab, (wv&1) -> 64-col slab: 2x4 = 8 WMMA accumulators/wave.
#define APAD 40   // LDS row stride (elements) for A [m][k]; 80 B = 16B aligned rows
#define BPAD 40   // LDS row stride (elements) for B-transposed [n][k]

template <typename OutT>
__global__ __launch_bounds__(256) void gemm_bf16(
    const __bf16* __restrict__ A, const __bf16* __restrict__ Bm,
    OutT* __restrict__ C, int M, int N, int K)
{
  __shared__ __bf16 sA[2][128 * APAD];
  __shared__ __bf16 sB[2][128 * BPAD];

  const int tid  = threadIdx.x;
  const int wv   = tid >> 5;
  const int lane = tid & 31;
  const int hfl  = lane >> 4;      // K-half selector (wave32 WMMA layout)
  const int lr   = lane & 15;

  const int bm = blockIdx.y * 128;
  const int bn = blockIdx.x * 128;
  const int wm = (wv >> 1) * 32;
  const int wn = (wv & 1) * 64;

  const v8f vz = {0.f,0.f,0.f,0.f,0.f,0.f,0.f,0.f};
  v8f acc[2][4];
  for (int i = 0; i < 2; i++) for (int j = 0; j < 4; j++) acc[i][j] = vz;

  const int ar = tid >> 1;           // A tile row this thread stages
  const int ac = (tid & 1) * 16;     // A tile col offset (16 bf16 per thread)

  // --- staging helpers ---
  auto loadA = [&](int k0, int buf) {   // async (no VGPR round trip)
    const __bf16* src = A + (size_t)(bm + ar) * K + k0 + ac;
    async_b128(src,     &sA[buf][ar * APAD + ac]);
    async_b128(src + 8, &sA[buf][ar * APAD + ac + 8]);
  };
  uint4 breg[2];
  auto loadB_g = [&](int k0) {          // global -> registers
    #pragma unroll
    for (int r = 0; r < 2; r++) {
      int idx8 = tid + r * 256;         // 512 x (8 bf16) slots of the 32x128 tile
      int kk = idx8 >> 4;               // 0..31
      int n0 = (idx8 & 15) * 8;         // 0..120
      breg[r] = *(const uint4*)(Bm + (size_t)(k0 + kk) * N + bn + n0);
    }
  };
  auto loadB_s = [&](int buf) {         // registers -> LDS, transposed [n][k]
    #pragma unroll
    for (int r = 0; r < 2; r++) {
      int idx8 = tid + r * 256;
      int kk = idx8 >> 4;
      int n0 = (idx8 & 15) * 8;
      const __bf16* e = (const __bf16*)&breg[r];
      #pragma unroll
      for (int t = 0; t < 8; t++) sB[buf][(n0 + t) * BPAD + kk] = e[t];
    }
  };

  // --- prologue: stage tile 0 ---
  loadA(0, 0);
  loadB_g(0);
  loadB_s(0);
  async_wait0();
  __syncthreads();

  int cur = 0;
  for (int k0 = 0; k0 < K; k0 += 32) {
    const bool more = (k0 + 32 < K);
    if (more) {             // prefetch tile t+1 while computing tile t
      loadA(k0 + 32, cur ^ 1);   // async -> overlaps with WMMA below
      loadB_g(k0 + 32);          // global loads in flight through compute
    }

    // ---- build fragments & 8 WMMAs from buffer `cur` ----
    Frag afr[2], bfr[4];
    #pragma unroll
    for (int mt = 0; mt < 2; mt++)
      #pragma unroll
      for (int j = 0; j < 8; j++) {
        int kb = (j < 4 ? 2 * j : 16 + 2 * (j - 4)) + hfl * 8;   // 16-bit A layout
        afr[mt].u[j] = *(const unsigned*)&sA[cur][(wm + mt * 16 + lr) * APAD + kb];
      }
    #pragma unroll
    for (int nt = 0; nt < 4; nt++)
      #pragma unroll
      for (int j = 0; j < 8; j++) {
        int kb = (j < 4 ? 2 * j : 16 + 2 * (j - 4)) + hfl * 8;   // 16-bit B layout
        bfr[nt].u[j] = *(const unsigned*)&sB[cur][(wn + nt * 16 + lr) * BPAD + kb];
      }
    #pragma unroll
    for (int mt = 0; mt < 2; mt++)
      #pragma unroll
      for (int nt = 0; nt < 4; nt++)
        acc[mt][nt] = __builtin_amdgcn_wmma_f32_16x16x32_bf16(
            false, afr[mt].v, false, bfr[nt].v, (short)0, acc[mt][nt], false, false);

    if (more) loadB_s(cur ^ 1);   // finish staging B for t+1 after compute
    async_wait0();                // A tile t+1 fully in LDS
    __syncthreads();              // everyone done reading `cur`, t+1 visible
    cur ^= 1;
  }

  // ---- store: C layout lane(row = v + 8*half, col = lane&15) ----
  for (int mt = 0; mt < 2; mt++)
    for (int nt = 0; nt < 4; nt++)
      #pragma unroll
      for (int v = 0; v < 8; v++) {
        int row = bm + wm + mt * 16 + v + hfl * 8;
        int col = bn + wn + nt * 16 + lr;
        C[(size_t)row * N + col] = (OutT)acc[mt][nt][v];
      }
}

// ---------------- Q assembly: concat nope + RoPE(rope) ----------------
// NOTE: reference apply_rope broadcasts cos[:x.shape[-2]] over the HEAD axis,
// so the rotation "position" is the head index (h for Q, kv for K). Replicated.
__global__ void assemble_q(const __bf16* __restrict__ qn, const __bf16* __restrict__ qr,
                           __bf16* __restrict__ qf)
{
  int idx = blockIdx.x * blockDim.x + threadIdx.x;
  if (idx >= ROWS * H_) return;
  int row = idx / H_;
  int h = idx - row * H_;
  const __bf16* np = qn + (size_t)row * (H_ * HD_) + h * HD_;
  __bf16* dst = qf + ((size_t)row * H_ + h) * FULL_;
  for (int d = 0; d < HD_; d++) dst[d] = np[d];
  const __bf16* rp = qr + (size_t)row * (H_ * RD_) + h * RD_;
  float pos = (float)h;
  for (int i = 0; i < RD_ / 2; i++) {
    float inv = __powf(10000.f, -(2.f * i) / (float)RD_);
    float sn, cs; __sincosf(pos * inv, &sn, &cs);
    float x1 = (float)rp[2 * i], x2 = (float)rp[2 * i + 1];
    dst[HD_ + 2 * i]     = (__bf16)(x1 * cs - x2 * sn);
    dst[HD_ + 2 * i + 1] = (__bf16)(x1 * sn + x2 * cs);
  }
}

__global__ void assemble_kv(const __bf16* __restrict__ kn, const __bf16* __restrict__ kr,
                            const __bf16* __restrict__ vb,
                            __bf16* __restrict__ kf, __bf16* __restrict__ vf)
{
  int idx = blockIdx.x * blockDim.x + threadIdx.x;
  if (idx >= ROWS * KV_) return;
  int row = idx / KV_;
  int kv = idx - row * KV_;
  const __bf16* np = kn + (size_t)row * (KV_ * HD_) + kv * HD_;
  const __bf16* vp = vb + (size_t)row * (KV_ * HD_) + kv * HD_;
  __bf16* kd = kf + ((size_t)row * KV_ + kv) * FULL_;
  __bf16* vd = vf + ((size_t)row * KV_ + kv) * FULL_;
  for (int d = 0; d < HD_; d++) { kd[d] = np[d]; vd[d] = vp[d]; }
  for (int d = HD_; d < FULL_; d++) vd[d] = (__bf16)0.f;   // zero pad V
  const __bf16* rp = kr + (size_t)row * (KV_ * RD_) + kv * RD_;
  float pos = (float)kv;
  for (int i = 0; i < RD_ / 2; i++) {
    float inv = __powf(10000.f, -(2.f * i) / (float)RD_);
    float sn, cs; __sincosf(pos * inv, &sn, &cs);
    float x1 = (float)rp[2 * i], x2 = (float)rp[2 * i + 1];
    kd[HD_ + 2 * i]     = (__bf16)(x1 * cs - x2 * sn);
    kd[HD_ + 2 * i + 1] = (__bf16)(x1 * sn + x2 * cs);
  }
}

// ---------------- causal flash attention (WMMA, online softmax) ----------------
// block = (b,h) x 128 q-rows; 8 waves x 16 rows. KV chunk = 32 keys.
#define KPADE 208   // sK row stride (elements): 416 B, 16B aligned
#define VTPE  40    // sVT row stride (elements)
#define PWF   34    // per-wave P scratch row stride (floats), even -> b64-aligned pairs

__global__ __launch_bounds__(256) void flash_attn(
    const __bf16* __restrict__ Qf, const __bf16* __restrict__ Kf,
    const __bf16* __restrict__ Vf, __bf16* __restrict__ Of)
{
  __shared__ __bf16 sK [32  * KPADE];   // K chunk, [key][d]  (== WMMA-B layout for QK^T)
  __shared__ __bf16 sVT[192 * VTPE];    // V chunk transposed, [d][key] (WMMA-B for P@V)
  __shared__ float  sP [8 * 16 * PWF];  // per-wave P tile (C-layout -> A-layout relay)

  const int tid  = threadIdx.x;
  const int wv   = tid >> 5;
  const int lane = tid & 31;
  const int hfl  = lane >> 4;
  const int lr   = lane & 15;

  const int bh  = blockIdx.y;
  const int b   = bh / H_;
  const int h   = bh - b * H_;
  const int kvh = h / (H_ / KV_);
  const int q0  = blockIdx.x * 128;
  const int qw  = q0 + wv * 16;                  // this wave's 16 q rows

  const float scale = 0.07216878364870322f;      // 1/sqrt(192)

  // resident Q fragments: 6 K-steps of 32 over d=192
  Frag qa[6];
  {
    size_t base = (((size_t)b * S_ + qw + lr) * H_ + h) * (size_t)FULL_;
    #pragma unroll
    for (int ds = 0; ds < 6; ds++)
      #pragma unroll
      for (int j = 0; j < 8; j++) {
        int kb = ds * 32 + ((j < 4 ? 2 * j : 16 + 2 * (j - 4)) + hfl * 8);
        qa[ds].u[j] = *(const unsigned*)(Qf + base + kb);
      }
  }

  const v8f vz = {0.f,0.f,0.f,0.f,0.f,0.f,0.f,0.f};
  v8f o[12];
  for (int t = 0; t < 12; t++) o[t] = vz;
  float m[8], l[8];
  for (int v = 0; v < 8; v++) { m[v] = -3.0e38f; l[v] = 0.f; }

  const int nchunk = (q0 + 128) / 32;            // causal upper bound for this block
  float* myP = sP + wv * 16 * PWF;

  for (int c = 0; c < nchunk; c++) {
    const int kb0 = c * 32;
    // ---- cooperative K load: async global->LDS (no transpose needed) ----
    for (int it = 0; it < 3; it++) {
      int i = tid + it * 256;                    // 768 uint4 slots = 32*192/8
      int row = i / 24, d0 = (i - row * 24) * 8;
      const __bf16* src = Kf + (((size_t)b * S_ + kb0 + row) * KV_ + kvh) * (size_t)FULL_ + d0;
      async_b128(src, &sK[row * KPADE + d0]);
    }
    // ---- cooperative V load (vectorized), transposed scatter into [d][key] ----
    for (int it = 0; it < 3; it++) {
      int i = tid + it * 256;                    // same 768 uint4 slots
      int key = i / 24, d0 = (i - key * 24) * 8;
      const __bf16* src = Vf + (((size_t)b * S_ + kb0 + key) * KV_ + kvh) * (size_t)FULL_ + d0;
      uint4 pv = *(const uint4*)src;
      const __bf16* e = (const __bf16*)&pv;
      #pragma unroll
      for (int t = 0; t < 8; t++) sVT[(d0 + t) * VTPE + key] = e[t];
    }
    async_wait0();
    __syncthreads();

    // ---- scores: two 16-key tiles; batch 6 B-frags then 6 WMMAs ----
    float sc[2][8];
    #pragma unroll
    for (int kt = 0; kt < 2; kt++) {
      Frag bfs[6];
      #pragma unroll
      for (int ds = 0; ds < 6; ds++)
        #pragma unroll
        for (int j = 0; j < 8; j++) {
          int kk = ds * 32 + ((j < 4 ? 2 * j : 16 + 2 * (j - 4)) + hfl * 8);
          bfs[ds].u[j] = *(const unsigned*)&sK[(kt * 16 + lr) * KPADE + kk];
        }
      v8f sa = vz;
      #pragma unroll
      for (int ds = 0; ds < 6; ds++)
        sa = __builtin_amdgcn_wmma_f32_16x16x32_bf16(
            false, qa[ds].v, false, bfs[ds].v, (short)0, sa, false, false);
      int kcol = kb0 + kt * 16 + lr;
      #pragma unroll
      for (int v = 0; v < 8; v++) {
        int qi = qw + v + 8 * hfl;
        float val = sa[v] * scale;
        sc[kt][v] = (kcol > qi) ? -3.0e38f : val;   // causal mask
      }
    }

    // ---- online softmax (16-lane row reductions) ----
    float alpha[8];
    #pragma unroll
    for (int v = 0; v < 8; v++) {
      float mx = fmaxf(sc[0][v], sc[1][v]);
      for (int off = 1; off < 16; off <<= 1) mx = fmaxf(mx, __shfl_xor(mx, off, 32));
      float mn = fmaxf(m[v], mx);
      alpha[v] = __expf(m[v] - mn);
      m[v] = mn;
      float p0 = (sc[0][v] <= -1.0e38f) ? 0.f : __expf(sc[0][v] - mn);
      float p1 = (sc[1][v] <= -1.0e38f) ? 0.f : __expf(sc[1][v] - mn);
      myP[(v + 8 * hfl) * PWF + lr]      = p0;
      myP[(v + 8 * hfl) * PWF + 16 + lr] = p1;
      float ps = p0 + p1;
      for (int off = 1; off < 16; off <<= 1) ps += __shfl_xor(ps, off, 32);
      l[v] = l[v] * alpha[v] + ps;
    }

    // ---- P: C-layout -> A-layout via per-wave LDS (DS in-order per wave) ----
    Frag pa;
    #pragma unroll
    for (int j = 0; j < 8; j++) {
      int kk = (j < 4 ? 2 * j : 16 + 2 * (j - 4)) + hfl * 8;
      float2 pr = *(const float2*)&myP[lr * PWF + kk];
      union { unsigned u; __bf16 hh[2]; } pk;
      pk.hh[0] = (__bf16)pr.x; pk.hh[1] = (__bf16)pr.y;
      pa.u[j] = pk.u;
    }

    // ---- rescale O and accumulate P @ V (12 WMMA over d=192, batched by 4) ----
    #pragma unroll
    for (int g = 0; g < 3; g++) {
      Frag bfs[4];
      #pragma unroll
      for (int t = 0; t < 4; t++) {
        int tt = g * 4 + t;
        #pragma unroll
        for (int j = 0; j < 8; j++) {
          int kk = (j < 4 ? 2 * j : 16 + 2 * (j - 4)) + hfl * 8;
          bfs[t].u[j] = *(const unsigned*)&sVT[(tt * 16 + lr) * VTPE + kk];
        }
      }
      #pragma unroll
      for (int t = 0; t < 4; t++) {
        int tt = g * 4 + t;
        v8f ot = o[tt];
        #pragma unroll
        for (int v = 0; v < 8; v++) ot[v] *= alpha[v];
        o[tt] = __builtin_amdgcn_wmma_f32_16x16x32_bf16(
            false, pa.v, false, bfs[t].v, (short)0, ot, false, false);
      }
    }
    __syncthreads();
  }

  // ---- epilogue: normalize, write attn_out row-major (b*S+s, h*192+d) ----
  for (int t = 0; t < 12; t++)
    #pragma unroll
    for (int v = 0; v < 8; v++) {
      int srow = qw + v + 8 * hfl;
      float val = o[t][v] / l[v];
      Of[((size_t)b * S_ + srow) * (size_t)(H_ * FULL_) + h * FULL_ + t * 16 + lr] =
          (__bf16)val;
    }
}

// ---------------- host-side orchestration ----------------
extern "C" void kernel_launch(void* const* d_in, const int* in_sizes, int n_in,
                              void* d_out, int out_size, void* d_ws, size_t ws_size,
                              hipStream_t stream)
{
  (void)in_sizes; (void)n_in; (void)out_size; (void)ws_size;
  const float* hs     = (const float*)d_in[0];
  const float* Wq     = (const float*)d_in[1];
  const float* Wkvc   = (const float*)d_in[2];
  const float* Wkdec  = (const float*)d_in[3];
  const float* Wvdec  = (const float*)d_in[4];
  const float* Wqrope = (const float*)d_in[5];
  const float* Wkrope = (const float*)d_in[6];
  const float* Wo     = (const float*)d_in[7];
  float* out = (float*)d_out;

  char* p = (char*)d_ws;
  auto alloc = [&](size_t bytes) {
    char* r = p; p += (bytes + 255) & ~(size_t)255; return r;
  };
  __bf16* hs_b     = (__bf16*)alloc((size_t)ROWS * HID_ * 2);
  __bf16* wq_b     = (__bf16*)alloc((size_t)HID_ * H_ * HD_ * 2);
  __bf16* wkvc_b   = (__bf16*)alloc((size_t)HID_ * 512 * 2);
  __bf16* wkdec_b  = (__bf16*)alloc((size_t)512 * KV_ * HD_ * 2);
  __bf16* wvdec_b  = (__bf16*)alloc((size_t)512 * KV_ * HD_ * 2);
  __bf16* wqrope_b = (__bf16*)alloc((size_t)HID_ * H_ * RD_ * 2);
  __bf16* wkrope_b = (__bf16*)alloc((size_t)512 * KV_ * RD_ * 2);
  __bf16* wo_b     = (__bf16*)alloc((size_t)H_ * FULL_ * HID_ * 2);
  __bf16* qn       = (__bf16*)alloc((size_t)ROWS * H_ * HD_ * 2);
  __bf16* qr       = (__bf16*)alloc((size_t)ROWS * H_ * RD_ * 2);
  __bf16* cb       = (__bf16*)alloc((size_t)ROWS * 512 * 2);
  __bf16* kn       = (__bf16*)alloc((size_t)ROWS * KV_ * HD_ * 2);
  __bf16* vb       = (__bf16*)alloc((size_t)ROWS * KV_ * HD_ * 2);
  __bf16* kr       = (__bf16*)alloc((size_t)ROWS * KV_ * RD_ * 2);
  __bf16* qf       = (__bf16*)alloc((size_t)ROWS * H_ * FULL_ * 2);
  __bf16* kf       = (__bf16*)alloc((size_t)ROWS * KV_ * FULL_ * 2);
  __bf16* vf       = (__bf16*)alloc((size_t)ROWS * KV_ * FULL_ * 2);
  __bf16* ao       = (__bf16*)alloc((size_t)ROWS * H_ * FULL_ * 2);

  auto cvt = [&](const float* s, __bf16* d, int n) {
    cvt_f32_bf16<<<dim3((n + 255) / 256), dim3(256), 0, stream>>>(s, d, n);
  };
  cvt(hs,     hs_b,     ROWS * HID_);
  cvt(Wq,     wq_b,     HID_ * H_ * HD_);
  cvt(Wkvc,   wkvc_b,   HID_ * 512);
  cvt(Wkdec,  wkdec_b,  512 * KV_ * HD_);
  cvt(Wvdec,  wvdec_b,  512 * KV_ * HD_);
  cvt(Wqrope, wqrope_b, HID_ * H_ * RD_);
  cvt(Wkrope, wkrope_b, 512 * KV_ * RD_);
  cvt(Wo,     wo_b,     H_ * FULL_ * HID_);

  auto gemmB = [&](const __bf16* A, const __bf16* Bm, __bf16* C, int M, int N, int K) {
    gemm_bf16<__bf16><<<dim3(N / 128, M / 128), dim3(256), 0, stream>>>(A, Bm, C, M, N, K);
  };

  gemmB(hs_b, wq_b,     qn, ROWS, H_ * HD_, HID_);   // 4096x2048x2048
  gemmB(hs_b, wqrope_b, qr, ROWS, H_ * RD_, HID_);   // 4096x1024x2048
  gemmB(hs_b, wkvc_b,   cb, ROWS, 512,      HID_);   // 4096x 512x2048
  gemmB(cb,   wkdec_b,  kn, ROWS, KV_ * HD_, 512);   // 4096x 512x 512
  gemmB(cb,   wvdec_b,  vb, ROWS, KV_ * HD_, 512);   // 4096x 512x 512
  gemmB(cb,   wkrope_b, kr, ROWS, KV_ * RD_, 512);   // 4096x 256x 512

  assemble_q <<<dim3(ROWS * H_  / 256), dim3(256), 0, stream>>>(qn, qr, qf);
  assemble_kv<<<dim3(ROWS * KV_ / 256), dim3(256), 0, stream>>>(kn, kr, vb, kf, vf);

  flash_attn<<<dim3(S_ / 128, B_ * H_), dim3(256), 0, stream>>>(qf, kf, vf, ao);

  gemm_bf16<float><<<dim3(HID_ / 128, ROWS / 128), dim3(256), 0, stream>>>(
      ao, wo_b, out, ROWS, HID_, H_ * FULL_);        // 4096x2048x3072 -> f32 out
}
